// DeltaNet_54932631715924
// MI455X (gfx1250) — matrix operand order repaired and checked by reading, not compiled, not run on hardware
//
#include <hip/hip_runtime.h>
#include <hip/hip_bf16.h>

#define BB   2
#define LL   4096
#define HIDD 1024
#define HH   4
#define DHH  256
#define KSS  4
#define CC   64
#define NCH  (LL / CC)
#define MTOK (BB * LL)
#define EPSV 1e-5f

typedef __bf16 bf16_t;
typedef __attribute__((ext_vector_type(8)))  __bf16 v8bf;
typedef __attribute__((ext_vector_type(16))) __bf16 v16bf;
typedef __attribute__((ext_vector_type(8)))  float  v8f;
typedef __attribute__((ext_vector_type(4)))  int    int4v;

#if defined(__has_builtin)
#if __has_builtin(__builtin_amdgcn_global_load_async_to_lds_b128)
#define HAVE_ASYNC_LDS 1
#endif
#endif

// 16B global -> LDS copy, ASYNCcnt-tracked on gfx1250; sync fallback otherwise
__device__ __forceinline__ void async_b128(const void* g, void* l) {
#ifdef HAVE_ASYNC_LDS
  __builtin_amdgcn_global_load_async_to_lds_b128((int4v*)g, (int4v*)l, 0, 0);
#else
  *(v8bf*)l = *(const v8bf*)g;
#endif
}
__device__ __forceinline__ void async_wait_all() {
#ifdef HAVE_ASYNC_LDS
#if __has_builtin(__builtin_amdgcn_s_wait_asynccnt)
  __builtin_amdgcn_s_wait_asynccnt(0);
#else
  asm volatile("s_wait_asynccnt 0x0" ::: "memory");
#endif
#endif
}
__device__ __forceinline__ void async_wait_keep1() {
#ifdef HAVE_ASYNC_LDS
#if __has_builtin(__builtin_amdgcn_s_wait_asynccnt)
  __builtin_amdgcn_s_wait_asynccnt(1);
#else
  asm volatile("s_wait_asynccnt 0x1" ::: "memory");
#endif
#endif
}

__device__ __forceinline__ bf16_t f2bf(float f) {
  unsigned u = __builtin_bit_cast(unsigned, f);
  unsigned r = (u + 0x7FFFu + ((u >> 16) & 1u)) >> 16;
  unsigned short h = (unsigned short)r;
  return __builtin_bit_cast(bf16_t, h);
}
__device__ __forceinline__ float bf2f(bf16_t b) {
  unsigned u = ((unsigned)__builtin_bit_cast(unsigned short, b)) << 16;
  return __builtin_bit_cast(float, u);
}
__device__ __forceinline__ float siluf(float x) { return x / (1.f + __expf(-x)); }

__device__ __forceinline__ v8f wmma_bf16(v16bf a, v16bf b, v8f c) {
  return __builtin_amdgcn_wmma_f32_16x16x32_bf16(false, a, false, b, (short)0, c,
                                                 false, false);
}

// ---- WMMA fragment helpers (A/B in contiguous k = 16*(lane/16) + i order, wave32) -------------------
__device__ __forceinline__ v16bf load_A_frag(const bf16_t* src, int ld, int m0, int k0) {
  int lane = threadIdx.x & 31;
  const bf16_t* p = src + (size_t)(m0 + (lane & 15)) * ld + k0 + ((lane >> 4) ? 16 : 0);
  v8bf lo = *(const v8bf*)p;
  v8bf hi = *(const v8bf*)(p + 8);
  v16bf a;
#pragma unroll
  for (int i = 0; i < 8; ++i) { a[i] = lo[i]; a[8 + i] = hi[i]; }
  return a;
}
__device__ __forceinline__ v16bf load_A_fragT(const bf16_t* src, int ld, int m0, int k0) {
  int lane = threadIdx.x & 31;
  int m = m0 + (lane & 15);
  int kb = k0 + ((lane >> 4) ? 16 : 0);
  v16bf a;
#pragma unroll
  for (int i = 0; i < 16; ++i) {
    int k = kb + i;
    a[i] = src[(size_t)k * ld + m];
  }
  return a;
}
__device__ __forceinline__ v16bf load_B_fragT(const bf16_t* src, int ld, int n0, int k0) {
  int lane = threadIdx.x & 31;
  const bf16_t* p = src + (size_t)(n0 + (lane & 15)) * ld + k0 + ((lane >> 4) ? 16 : 0);
  v8bf lo = *(const v8bf*)p;
  v8bf hi = *(const v8bf*)(p + 8);
  v16bf b;
#pragma unroll
  for (int i = 0; i < 8; ++i) { b[i] = lo[i]; b[8 + i] = hi[i]; }
  return b;
}
__device__ __forceinline__ v16bf load_B_frag(const bf16_t* src, int ld, int k0, int n0) {
  int lane = threadIdx.x & 31;
  int n = n0 + (lane & 15);
  int kb = k0 + ((lane >> 4) ? 16 : 0);
  v16bf b;
#pragma unroll
  for (int i = 0; i < 16; ++i) b[i] = src[(size_t)(kb + i) * ld + n];
  return b;
}
__device__ __forceinline__ void store_C(float* dst, int ld, int m0, int n0, v8f c) {
  int lane = threadIdx.x & 31;
  int nn = n0 + (lane & 15);
  int mb = m0 + ((lane >> 4) ? 8 : 0);
#pragma unroll
  for (int v = 0; v < 8; ++v) dst[(size_t)(mb + v) * ld + nn] = c[v];
}
__device__ __forceinline__ v8f load_C(const float* src, int ld, int m0, int n0) {
  int lane = threadIdx.x & 31;
  int nn = n0 + (lane & 15);
  int mb = m0 + ((lane >> 4) ? 8 : 0);
  v8f c;
#pragma unroll
  for (int v = 0; v < 8; ++v) c[v] = src[(size_t)(mb + v) * ld + nn];
  return c;
}

// ---- fp32 -> bf16 convert -------------------------------------------------
__global__ void cvt_f32_bf16(const float* __restrict__ in, bf16_t* __restrict__ out, int n) {
  int i = blockIdx.x * blockDim.x + threadIdx.x;
  int stride = gridDim.x * blockDim.x;
  for (; i < n; i += stride) out[i] = f2bf(in[i]);
}

// ---- Y = X @ W^T : LDS-staged weight slab (async), pipelined A loads ------
__global__ void __launch_bounds__(256) gemm_xwT_bf16(const bf16_t* __restrict__ X,
                                                     const bf16_t* __restrict__ W,
                                                     float* __restrict__ Y,
                                                     int M, int N, int K) {
  __shared__ bf16_t sW[2][64 * 32];  // double-buffered 64N x 32K slab, 8KB
  const int tid = threadIdx.x;
  const int wave = tid >> 5;
  const int m0 = blockIdx.y * 128 + wave * 16;
  const int n0 = blockIdx.x * 64;

  // one b128 async op per thread per slab (exactly 1 instr per wave)
  const int srow = tid >> 2;       // 0..63
  const int sseg = (tid & 3) * 8;  // bf16 offset within 32-wide k slab

  v8f acc0 = {}, acc1 = {}, acc2 = {}, acc3 = {};
  const int nk = K / 32;

  async_b128(W + (size_t)(n0 + srow) * K + sseg, &sW[0][srow * 32 + sseg]);

  v16bf a = load_A_frag(X, K, m0, 0);
  for (int ki = 0; ki < nk; ++ki) {
    const int k0 = ki * 32;
    if (ki + 1 < nk) {
      async_b128(W + (size_t)(n0 + srow) * K + k0 + 32 + sseg,
                 &sW[(ki + 1) & 1][srow * 32 + sseg]);
      async_wait_keep1();
    } else {
      async_wait_all();
    }
    __syncthreads();
    // issue next A fragment before this tile's WMMAs (pipelining)
    v16bf an;
    if (ki + 1 < nk) {
      __builtin_prefetch(X + (size_t)(m0 + (tid & 15)) * K + k0 + 64, 0, 1);
      an = load_A_frag(X, K, m0, k0 + 32);
    }
    const bf16_t* Wl = sW[ki & 1];
    v16bf b0 = load_B_fragT(Wl, 32, 0, 0);
    v16bf b1 = load_B_fragT(Wl, 32, 16, 0);
    v16bf b2 = load_B_fragT(Wl, 32, 32, 0);
    v16bf b3 = load_B_fragT(Wl, 32, 48, 0);
    acc0 = wmma_bf16(a, b0, acc0);
    acc1 = wmma_bf16(a, b1, acc1);
    acc2 = wmma_bf16(a, b2, acc2);
    acc3 = wmma_bf16(a, b3, acc3);
    if (ki + 1 < nk) a = an;
    __syncthreads();
  }
  store_C(Y, N, m0, n0 +  0, acc0);
  store_C(Y, N, m0, n0 + 16, acc1);
  store_C(Y, N, m0, n0 + 32, acc2);
  store_C(Y, N, m0, n0 + 48, acc3);
}

// ---- beta = sigmoid(x @ Wbeta^T), one block per token ---------------------
__global__ void __launch_bounds__(128) beta_kernel(const float* __restrict__ x,
                                                   const float* __restrict__ Wb,
                                                   float* __restrict__ beta) {
  __shared__ float acc[HH];
  const int g = blockIdx.x;
  const int tid = threadIdx.x;
  if (tid < HH) acc[tid] = 0.f;
  __syncthreads();
  float p0 = 0, p1 = 0, p2 = 0, p3 = 0;
  const float* xr = x + (size_t)g * HIDD;
  for (int k = tid; k < HIDD; k += 128) {
    float xv = xr[k];
    p0 += xv * Wb[0 * HIDD + k];
    p1 += xv * Wb[1 * HIDD + k];
    p2 += xv * Wb[2 * HIDD + k];
    p3 += xv * Wb[3 * HIDD + k];
  }
  atomicAdd(&acc[0], p0); atomicAdd(&acc[1], p1);
  atomicAdd(&acc[2], p2); atomicAdd(&acc[3], p3);
  __syncthreads();
  if (tid < HH) beta[(size_t)g * HH + tid] = 1.f / (1.f + __expf(-acc[tid]));
}

// ---- conv(KS=4,causal) + silu + silu (+L2 norm), bf16 out, optional beta --
// mode bit0: write raw bf16; mode bit1: write beta-scaled bf16
__global__ void __launch_bounds__(256) conv_silu_kernel(const float* __restrict__ lin,
                                                        const float* __restrict__ w,
                                                        const float* __restrict__ beta,
                                                        bf16_t* __restrict__ out_raw,
                                                        bf16_t* __restrict__ out_scl,
                                                        int mode, int do_norm) {
  __shared__ float ssq[HH];
  const int g = blockIdx.x;
  const int b = g / LL, t = g % LL;
  const int tid = threadIdx.x;
  if (tid < HH) ssq[tid] = 0.f;
  __syncthreads();
  const int c0 = tid * 4;
  const int h = c0 >> 8;
  float y[4];
  float sum2 = 0.f;
#pragma unroll
  for (int j = 0; j < 4; ++j) {
    int c = c0 + j;
    float acc = 0.f;
#pragma unroll
    for (int s = 0; s < KSS; ++s) {
      int tt = t - (KSS - 1) + s;
      if (tt >= 0) acc += w[c * KSS + s] * lin[(size_t)(b * LL + tt) * HIDD + c];
    }
    acc = siluf(siluf(acc));
    y[j] = acc;
    sum2 += acc * acc;
  }
  float inv = 1.f;
  if (do_norm) {
    atomicAdd(&ssq[h], sum2);
    __syncthreads();
    inv = 1.f / fmaxf(sqrtf(ssq[h]), 1e-12f);
  }
  float bt = beta[(size_t)g * HH + h];
#pragma unroll
  for (int j = 0; j < 4; ++j) {
    float v = y[j] * inv;
    if (mode & 1) out_raw[(size_t)g * HIDD + c0 + j] = f2bf(v);
    if (mode & 2) out_scl[(size_t)g * HIDD + c0 + j] = f2bf(v * bt);
  }
}

// ---- chunked delta rule: 1 block per (b,h), async LDS staging -------------
__global__ void __launch_bounds__(256) delta_rule_kernel(const bf16_t* __restrict__ qb,
                                                         const bf16_t* __restrict__ kb,
                                                         const bf16_t* __restrict__ kbb,
                                                         const bf16_t* __restrict__ vbb,
                                                         float* __restrict__ obuf,
                                                         float* __restrict__ Sout) {
  extern __shared__ char smem[];
  bf16_t* sQ  = (bf16_t*)smem;            // 64x256
  bf16_t* sK  = sQ  + CC * DHH;           // raw K
  bf16_t* sKb = sK  + CC * DHH;           // K*beta
  bf16_t* sVb = sKb + CC * DHH;           // V*beta
  bf16_t* sU  = sVb + CC * DHH;           // u
  float*  sT  = (float*)(sU + CC * DHH);  // 64x64 f32 scratch
  bf16_t* sTb = (bf16_t*)(sT + CC * CC);
  bf16_t* sPb = sTb + CC * CC;
  float*  dS  = (float*)(sPb + CC * CC);  // 256 diag(S)

  const int tid  = threadIdx.x;
  const int lane = tid & 31;
  const int wave = tid >> 5;
  const int bh = blockIdx.x;
  const int bb = bh / HH, hh = bh % HH;
  float* S_g = Sout + (size_t)bh * DHH * DHH;

  dS[tid] = 0.f;
  __syncthreads();

  for (int n = 0; n < NCH; ++n) {
    const int t0 = n * CC;
    // async-stage the 4 chunk operands (4 x 32KB, b128 granules)
    for (int i = tid; i < CC * 32; i += 256) {  // 2048 segs of 8 bf16
      int c = i >> 5;
      int e = (i & 31) * 8;
      size_t g = (size_t)(bb * LL + t0 + c) * HIDD + hh * DHH + e;
      int o = c * DHH + e;
      async_b128(qb  + g, sQ  + o);
      async_b128(kb  + g, sK  + o);
      async_b128(kbb + g, sKb + o);
      async_b128(vbb + g, sVb + o);
    }
    async_wait_all();
    __syncthreads();

    // A = Kb @ K^T  (64x64, reduce 256) -> sT
    for (int tile = wave; tile < 16; tile += 8) {
      int tm = (tile >> 2) << 4, tn = (tile & 3) << 4;
      v8f acc = {};
#pragma unroll
      for (int k0 = 0; k0 < DHH; k0 += 32) {
        v16bf a = load_A_frag(sKb, DHH, tm, k0);
        v16bf b = load_B_fragT(sK, DHH, tn, k0);
        acc = wmma_bf16(a, b, acc);
      }
      store_C(sT, CC, tm, tn, acc);
    }
    __syncthreads();
    // T = -tril(A,-1) + I
    for (int i = tid; i < CC * CC; i += 256) {
      int r = i >> 6, cj = i & 63;
      float v = sT[i];
      sT[i] = (cj < r) ? -v : (cj == r ? 1.f : 0.f);
    }
    __syncthreads();
    // forward substitution
    for (int i = 1; i < CC; ++i) {
      float prod = 0.f;
      if (tid < CC) {
        for (int j = 0; j < CC; ++j) prod += sT[i * CC + j] * sT[j * CC + tid];
      }
      __syncthreads();
      if (tid < i) sT[i * CC + tid] += prod;
      __syncthreads();
    }
    for (int i = tid; i < CC * CC; i += 256) sTb[i] = f2bf(sT[i]);
    __syncthreads();

    // P = tril(Q @ K^T) -> sPb (reuse sT)
    for (int tile = wave; tile < 16; tile += 8) {
      int tm = (tile >> 2) << 4, tn = (tile & 3) << 4;
      v8f acc = {};
#pragma unroll
      for (int k0 = 0; k0 < DHH; k0 += 32) {
        v16bf a = load_A_frag(sQ, DHH, tm, k0);
        v16bf b = load_B_fragT(sK, DHH, tn, k0);
        acc = wmma_bf16(a, b, acc);
      }
      store_C(sT, CC, tm, tn, acc);
    }
    __syncthreads();
    for (int i = tid; i < CC * CC; i += 256) {
      int r = i >> 6, cj = i & 63;
      sPb[i] = f2bf((cj <= r) ? sT[i] : 0.f);
    }
    __syncthreads();

    // u = T@Vb - (T@Kb) * diag(S)
    for (int tile = wave; tile < 64; tile += 8) {
      int tm = (tile >> 4) << 4, tn = (tile & 15) << 4;
      v8f aU = {}, aW = {};
#pragma unroll
      for (int k0 = 0; k0 < CC; k0 += 32) {
        v16bf a  = load_A_frag(sTb, CC, tm, k0);
        v16bf bu = load_B_frag(sVb, DHH, k0, tn);
        v16bf bw = load_B_frag(sKb, DHH, k0, tn);
        aU = wmma_bf16(a, bu, aU);
        aW = wmma_bf16(a, bw, aW);
      }
      int nn = tn + (lane & 15);
      int mb = tm + ((lane >> 4) ? 8 : 0);
      float ds = dS[nn];
#pragma unroll
      for (int v = 0; v < 8; ++v)
        sU[(size_t)(mb + v) * DHH + nn] = f2bf(aU[v] - aW[v] * ds);
    }
    __syncthreads();

    // o = P @ u + Q * diag(S) -> global
    for (int tile = wave; tile < 64; tile += 8) {
      int tm = (tile >> 4) << 4, tn = (tile & 15) << 4;
      v8f acc = {};
#pragma unroll
      for (int k0 = 0; k0 < CC; k0 += 32) {
        v16bf a = load_A_frag(sPb, CC, tm, k0);
        v16bf b = load_B_frag(sU, DHH, k0, tn);
        acc = wmma_bf16(a, b, acc);
      }
      int nn = tn + (lane & 15);
      int mb = tm + ((lane >> 4) ? 8 : 0);
      float ds = dS[nn];
#pragma unroll
      for (int v = 0; v < 8; ++v) {
        int c = mb + v;
        float oi = bf2f(sQ[(size_t)c * DHH + nn]) * ds;
        obuf[(size_t)(bb * LL + t0 + c) * HIDD + hh * DHH + nn] = acc[v] + oi;
      }
    }

    // S += K^T @ u  (256x256, reduce 64), WMMA C-operand accumulation
    for (int tile = wave; tile < 256; tile += 8) {
      int tm = (tile >> 4) << 4, tn = (tile & 15) << 4;
      v8f acc = {};
      if (n) acc = load_C(S_g, DHH, tm, tn);
#pragma unroll
      for (int k0 = 0; k0 < CC; k0 += 32) {
        v16bf a = load_A_fragT(sK, DHH, tm, k0);  // A[d][c] = K[c][d]
        v16bf b = load_B_frag(sU, DHH, k0, tn);
        acc = wmma_bf16(a, b, acc);
      }
      store_C(S_g, DHH, tm, tn, acc);
    }
    __syncthreads();
    // diag(S) incremental update
    {
      float s = 0.f;
      int d = tid;
      for (int c = 0; c < CC; ++c)
        s += bf2f(sK[(size_t)c * DHH + d]) * bf2f(sU[(size_t)c * DHH + d]);
      dS[d] += s;
    }
    __syncthreads();
  }
}

// ---- RMSNorm + scale, bf16 out for final GEMM -----------------------------
__global__ void __launch_bounds__(256) rmsnorm_kernel(const float* __restrict__ o,
                                                      const float* __restrict__ w,
                                                      bf16_t* __restrict__ ob) {
  __shared__ float ssum;
  const int g = blockIdx.x;
  const int tid = threadIdx.x;
  if (tid == 0) ssum = 0.f;
  __syncthreads();
  float p = 0.f;
  const float* orow = o + (size_t)g * HIDD;
  for (int k = tid; k < HIDD; k += 256) { float v = orow[k]; p += v * v; }
  atomicAdd(&ssum, p);
  __syncthreads();
  float inv = rsqrtf(ssum * (1.f / HIDD) + EPSV);
  for (int k = tid; k < HIDD; k += 256)
    ob[(size_t)g * HIDD + k] = f2bf(orow[k] * inv * w[k]);
}

extern "C" void kernel_launch(void* const* d_in, const int* in_sizes, int n_in,
                              void* d_out, int out_size, void* d_ws, size_t ws_size,
                              hipStream_t stream) {
  (void)in_sizes; (void)n_in; (void)out_size; (void)ws_size;
  const float* x     = (const float*)d_in[0];
  const float* Wq    = (const float*)d_in[1];
  const float* Wk    = (const float*)d_in[2];
  const float* Wv    = (const float*)d_in[3];
  const float* convq = (const float*)d_in[4];
  const float* convk = (const float*)d_in[5];
  const float* convv = (const float*)d_in[6];
  const float* Wbeta = (const float*)d_in[7];
  const float* rmsw  = (const float*)d_in[8];
  const float* Wo    = (const float*)d_in[9];

  char* ws = (char*)d_ws;
  auto alloc = [&](size_t bytes) -> char* {
    char* p = ws;
    ws += (bytes + 255) & ~(size_t)255;
    return p;
  };
  const size_t NTH = (size_t)MTOK * HIDD;  // 8,388,608
  bf16_t* xb   = (bf16_t*)alloc(NTH * 2);
  bf16_t* Wqb  = (bf16_t*)alloc((size_t)HIDD * HIDD * 2);
  bf16_t* Wkb  = (bf16_t*)alloc((size_t)HIDD * HIDD * 2);
  bf16_t* Wvb  = (bf16_t*)alloc((size_t)HIDD * HIDD * 2);
  bf16_t* Wob  = (bf16_t*)alloc((size_t)HIDD * HIDD * 2);
  float*  lin  = (float*)alloc(NTH * 4);
  bf16_t* qbuf = (bf16_t*)alloc(NTH * 2);   // q (raw)
  bf16_t* kbuf = (bf16_t*)alloc(NTH * 2);   // k (raw)
  bf16_t* kbb  = (bf16_t*)alloc(NTH * 2);   // k*beta
  bf16_t* vbb  = (bf16_t*)alloc(NTH * 2);   // v*beta
  float*  betb = (float*)alloc((size_t)MTOK * HH * 4);
  float*  obuf = (float*)alloc(NTH * 4);
  bf16_t* ormb = (bf16_t*)alloc(NTH * 2);

  float* out_o = (float*)d_out;
  float* out_S = out_o + NTH;  // (B,H,DH,DH) region

  cvt_f32_bf16<<<1024, 256, 0, stream>>>(x,  xb,  (int)NTH);
  cvt_f32_bf16<<<256,  256, 0, stream>>>(Wq, Wqb, HIDD * HIDD);
  cvt_f32_bf16<<<256,  256, 0, stream>>>(Wk, Wkb, HIDD * HIDD);
  cvt_f32_bf16<<<256,  256, 0, stream>>>(Wv, Wvb, HIDD * HIDD);
  cvt_f32_bf16<<<256,  256, 0, stream>>>(Wo, Wob, HIDD * HIDD);

  beta_kernel<<<MTOK, 128, 0, stream>>>(x, Wbeta, betb);

  dim3 ggrid(HIDD / 64, MTOK / 128);
  gemm_xwT_bf16<<<ggrid, 256, 0, stream>>>(xb, Wqb, lin, MTOK, HIDD, HIDD);
  conv_silu_kernel<<<MTOK, 256, 0, stream>>>(lin, convq, betb, qbuf, qbuf, 1, 1);
  gemm_xwT_bf16<<<ggrid, 256, 0, stream>>>(xb, Wkb, lin, MTOK, HIDD, HIDD);
  conv_silu_kernel<<<MTOK, 256, 0, stream>>>(lin, convk, betb, kbuf, kbb, 3, 1);
  gemm_xwT_bf16<<<ggrid, 256, 0, stream>>>(xb, Wvb, lin, MTOK, HIDD, HIDD);
  conv_silu_kernel<<<MTOK, 256, 0, stream>>>(lin, convv, betb, vbb, vbb, 2, 0);

  const size_t dls = (size_t)(5 * CC * DHH * 2)   // sQ,sK,sKb,sVb,sU
                   + (size_t)(CC * CC * 4)        // sT
                   + (size_t)(CC * CC * 2) * 2    // sTb, sPb
                   + (size_t)(DHH * 4);           // dS
  delta_rule_kernel<<<BB * HH, 256, dls, stream>>>(qbuf, kbuf, kbb, vbb, obuf, out_S);

  rmsnorm_kernel<<<MTOK, 256, 0, stream>>>(obuf, rmsw, ormb);
  gemm_xwT_bf16<<<ggrid, 256, 0, stream>>>(ormb, Wob, out_o, MTOK, HIDD, HIDD);
}
